// vaRNN_76656576299372
// MI455X (gfx1250) — matrix operand order repaired and checked by reading, not compiled, not run on hardware
//
#include <hip/hip_runtime.h>

// ---------------------------------------------------------------------------
// MI455X (gfx1250) RNN-LM decode loop.
//   VOCAB=50257, EMBED=512, HIDDEN=1024, SEQ=512, batch=1, strictly sequential.
// bf16 weights (~120MB) stay resident in the 192MB L2; per step: one fused
// LSTM-cell kernel (x-gather folded in, gates split across waves) and one
// logits+argmax kernel.  All mat-vecs via v_wmma_f32_16x16x32_bf16.
// ---------------------------------------------------------------------------

#define VOCAB   50257
#define EMBED   512
#define HIDDEN  1024
#define SEQLEN  512
#define GATES   (4 * HIDDEN)   // 4096
#define XDIM    (2 * EMBED)    // 1024

typedef __attribute__((ext_vector_type(16))) __bf16        v16bf;
typedef __attribute__((ext_vector_type(8)))  float         v8f;
typedef __attribute__((ext_vector_type(4)))  unsigned int  v4u;

union BF16x16 { v16bf v; v4u u[2]; };

__device__ __forceinline__ unsigned short f32_to_bf16(float f) {
    unsigned int b = __float_as_uint(f);
    unsigned int r = b + 0x7FFFu + ((b >> 16) & 1u);   // round-to-nearest-even
    return (unsigned short)(r >> 16);
}

// A operand for a batch-1 mat-vec: logical A is 16x32, only row 0 valid
// (= x[k0 .. k0+31]).  Contiguous K order matching B: lane 0 holds K={k0..+15},
// lane 16 holds K={k0+16..+31}; all other lanes zero (selects only,
// EXEC stays all-ones as WMMA requires).
template <typename PT>
__device__ __forceinline__ BF16x16 build_vec_A(PT x, int k0, int lane) {
    const int olane = (lane >= 16) ? 16 : 0;
    const v4u l0 = *(const v4u*)(x + k0 + olane);
    const v4u l1 = *(const v4u*)(x + k0 + olane + 8);
    const v4u z  = {0u, 0u, 0u, 0u};
    const bool act = ((lane & 15) == 0);     // lanes 0 and 16 carry row 0
    BF16x16 a;
    a.u[0] = act ? l0 : z;
    a.u[1] = act ? l1 : z;
    return a;
}

// One 16-output mat-vec partial: acc(row0) += W[rows] . x, K = 1024.
template <typename PT>
__device__ __forceinline__ v8f matvec16(PT As, const unsigned short* __restrict__ wr0,
                                        int lane) {
    v8f acc = {};
    for (int k0 = 0; k0 < 1024; k0 += 32) {
        const BF16x16 a = build_vec_A(As, k0, lane);
        BF16x16 b;
        b.u[0] = *(const v4u*)(wr0 + k0);
        b.u[1] = *(const v4u*)(wr0 + k0 + 8);
        acc = __builtin_amdgcn_wmma_f32_16x16x32_bf16(
            false, a.v, false, b.v, (short)0, acc, false, false);
    }
    return acc;
}

// ---------------------------------------------------------------------------
// One-time f32 -> bf16 weight conversion (vectorized 4-wide, grid-stride).
// ---------------------------------------------------------------------------
__global__ void cvt_bf16_kernel(const float* __restrict__ s,
                                unsigned short* __restrict__ d,
                                long long n4) {
    long long i      = (long long)blockIdx.x * blockDim.x + threadIdx.x;
    long long stride = (long long)gridDim.x * blockDim.x;
    for (; i < n4; i += stride) {
        const float4 f = ((const float4*)s)[i];
        ushort4 r;
        r.x = f32_to_bf16(f.x); r.y = f32_to_bf16(f.y);
        r.z = f32_to_bf16(f.z); r.w = f32_to_bf16(f.w);
        ((ushort4*)d)[i] = r;
    }
}

__global__ void init_state_kernel(const float* __restrict__ h0,
                                  const float* __restrict__ c0,
                                  unsigned short* __restrict__ hbuf0,
                                  float* __restrict__ cst,
                                  unsigned long long* __restrict__ slots) {
    const int i = blockIdx.x * blockDim.x + threadIdx.x;
    if (i < HIDDEN) { hbuf0[i] = f32_to_bf16(h0[i]); cst[i] = c0[i]; }
    if (i < 2) slots[i] = 0ull;
}

// ---------------------------------------------------------------------------
// Fused LSTM cell.  Grid: 64 blocks x 256 thr (512 waves total).
// Block b owns hidden tile j in [16b, 16b+16).  Each block first rebuilds
// x_t = concat(emb[argmax_{t-1}], emb[tok_t]) into LDS (redundant per block;
// this removes a whole kernel launch per step).  Wave w computes gate (w&3)
// for part (w>>2): part 0 = x @ w_ih.T from LDS, part 1 = h @ w_hh.T from
// global.  Partials are combined through LDS; wave 0 applies the gate
// nonlinearities and updates c (f32) and h (bf16, double-buffered).
// ---------------------------------------------------------------------------
__global__ void lstm_cell_kernel(const unsigned short* __restrict__ wih,
                                 const unsigned short* __restrict__ whh,
                                 const int* __restrict__ seq,
                                 const float* __restrict__ emb,
                                 const unsigned short* __restrict__ hin,
                                 unsigned short* __restrict__ hout,
                                 float* __restrict__ cst,
                                 const float* __restrict__ b_ih,
                                 const float* __restrict__ b_hh,
                                 const unsigned long long* __restrict__ prev_slot,
                                 unsigned long long* __restrict__ cur_slot,
                                 int t) {
    __shared__ __align__(16) unsigned short xs[XDIM];
    __shared__ float psum[8][16];
    const int tid  = threadIdx.x;
    const int lane = tid & 31;
    const int w    = tid >> 5;      // wave in block: 0..7
    const int part = w >> 2;        // 0: x@w_ih, 1: h@w_hh
    const int gate = w & 3;         // i,f,g,o
    const int j0   = blockIdx.x * 16;

    if (tid == 0 && blockIdx.x == 0) *cur_slot = 0ull;   // reset this step's slot

    // Cooperative gather of x_t into LDS (4 elements per thread).
    {
        const int tok = seq[t];
        unsigned int idx = 0u;
        if (t > 0) idx = ~((unsigned int)(*prev_slot));  // packed low bits = ~argmax
        const int e0 = tid * 4;
#pragma unroll
        for (int q = 0; q < 4; ++q) {
            const int e = e0 + q;
            float v;
            if (e < EMBED) v = (t > 0) ? emb[(size_t)idx * EMBED + e] : 0.0f;
            else           v = emb[(size_t)tok * EMBED + (e - EMBED)];
            xs[e] = f32_to_bf16(v);
        }
    }
    __syncthreads();

    const int koff = (lane >= 16) ? 16 : 0;
    const int row  = gate * HIDDEN + j0 + (lane & 15);   // B column n = weight row
    v8f acc;
    if (part == 0) {   // wave-uniform branch; EXEC stays all-ones inside
        acc = matvec16((const unsigned short*)xs,
                       wih + (size_t)row * XDIM + koff, lane);
    } else {
        acc = matvec16(hin, whh + (size_t)row * HIDDEN + koff, lane);
    }
    if (lane < 16) psum[w][lane] = acc[0];
    __syncthreads();

    if (w == 0 && lane < 16) {
        const int j  = j0 + lane;
        const float gi = psum[0][lane] + psum[4][lane] + b_ih[j]              + b_hh[j];
        const float gf = psum[1][lane] + psum[5][lane] + b_ih[HIDDEN + j]     + b_hh[HIDDEN + j];
        const float gg = psum[2][lane] + psum[6][lane] + b_ih[2 * HIDDEN + j] + b_hh[2 * HIDDEN + j];
        const float go = psum[3][lane] + psum[7][lane] + b_ih[3 * HIDDEN + j] + b_hh[3 * HIDDEN + j];
        const float i_ = 1.0f / (1.0f + __expf(-gi));
        const float f_ = 1.0f / (1.0f + __expf(-gf));
        const float g_ = tanhf(gg);
        const float o_ = 1.0f / (1.0f + __expf(-go));
        const float c2 = f_ * cst[j] + i_ * g_;
        cst[j]  = c2;
        hout[j] = f32_to_bf16(o_ * tanhf(c2));
    }
}

// ---------------------------------------------------------------------------
// logits = h @ w_lin.T + b_lin over VOCAB, fused argmax.
// Each wave computes 32 vocab rows as two B tiles sharing one A build
// (2 independent WMMA chains -> better ILP, half the A overhead).
// Block-reduce of packed (monotone-float-key | ~index), one atomicMax/block.
// Grid: 197 blocks x 256 thr = 1576 waves >= ceil(50257/32).
// ---------------------------------------------------------------------------
__global__ void logits_kernel(const unsigned short* __restrict__ wlin,
                              const unsigned short* __restrict__ hb,
                              const float* __restrict__ b_lin,
                              unsigned long long* __restrict__ slot,
                              float* __restrict__ out_or_null) {
    __shared__ unsigned long long red[256];
    const int lane = threadIdx.x & 31;
    const int gw   = (blockIdx.x * blockDim.x + threadIdx.x) >> 5;
    const int n0   = gw * 32;
    const int koff = (lane >= 16) ? 16 : 0;
    int r0 = n0 + (lane & 15);      if (r0 > VOCAB - 1) r0 = VOCAB - 1;
    int r1 = n0 + 16 + (lane & 15); if (r1 > VOCAB - 1) r1 = VOCAB - 1;
    const unsigned short* wr0 = wlin + (size_t)r0 * HIDDEN + koff;
    const unsigned short* wr1 = wlin + (size_t)r1 * HIDDEN + koff;

    v8f acc0 = {}, acc1 = {};
    for (int k0 = 0; k0 < HIDDEN; k0 += 32) {
        const BF16x16 a = build_vec_A(hb, k0, lane);
        BF16x16 b0, b1;
        b0.u[0] = *(const v4u*)(wr0 + k0);
        b0.u[1] = *(const v4u*)(wr0 + k0 + 8);
        b1.u[0] = *(const v4u*)(wr1 + k0);
        b1.u[1] = *(const v4u*)(wr1 + k0 + 8);
        acc0 = __builtin_amdgcn_wmma_f32_16x16x32_bf16(
            false, a.v, false, b0.v, (short)0, acc0, false, false);
        acc1 = __builtin_amdgcn_wmma_f32_16x16x32_bf16(
            false, a.v, false, b1.v, (short)0, acc1, false, false);
    }

    unsigned long long key = 0ull;
    if (lane < 16) {
        const int v0 = n0 + lane;
        const int v1 = n0 + 16 + lane;
        if (v0 < VOCAB) {
            const float logit = acc0[0] + b_lin[v0];
            if (out_or_null) out_or_null[v0] = logit;
            const unsigned int fb = __float_as_uint(logit);
            const unsigned int mk = (fb & 0x80000000u) ? ~fb : (fb | 0x80000000u);
            key = ((unsigned long long)mk << 32) |
                  (unsigned long long)(~(unsigned int)v0);
        }
        if (v1 < VOCAB) {
            const float logit = acc1[0] + b_lin[v1];
            if (out_or_null) out_or_null[v1] = logit;
            const unsigned int fb = __float_as_uint(logit);
            const unsigned int mk = (fb & 0x80000000u) ? ~fb : (fb | 0x80000000u);
            const unsigned long long k1 =
                ((unsigned long long)mk << 32) |
                (unsigned long long)(~(unsigned int)v1);
            if (k1 > key) key = k1;
        }
    }
    red[threadIdx.x] = key;
    __syncthreads();
    for (int s = 128; s > 0; s >>= 1) {
        if ((int)threadIdx.x < s) {
            if (red[threadIdx.x + s] > red[threadIdx.x])
                red[threadIdx.x] = red[threadIdx.x + s];
        }
        __syncthreads();
    }
    if (threadIdx.x == 0) atomicMax(slot, red[0]);
}

// ---------------------------------------------------------------------------
// Host launcher: 3 conversion kernels + init, then 512 x 2 step kernels.
// All launches on `stream` (graph-capture safe, deterministic).
// ---------------------------------------------------------------------------
extern "C" void kernel_launch(void* const* d_in, const int* in_sizes, int n_in,
                              void* d_out, int out_size, void* d_ws, size_t ws_size,
                              hipStream_t stream) {
    const int*   seq   = (const int*)  d_in[0];
    const float* h0    = (const float*)d_in[1];
    const float* c0    = (const float*)d_in[2];
    const float* emb   = (const float*)d_in[3];
    const float* w_ih  = (const float*)d_in[4];
    const float* w_hh  = (const float*)d_in[5];
    const float* b_ih  = (const float*)d_in[6];
    const float* b_hh  = (const float*)d_in[7];
    const float* w_lin = (const float*)d_in[8];
    const float* b_lin = (const float*)d_in[9];

    // Workspace layout (all sub-offsets 16B aligned):
    unsigned short* wihb  = (unsigned short*)d_ws;
    unsigned short* whhb  = wihb  + (size_t)GATES * XDIM;        //  8.39 MB in
    unsigned short* wlinb = whhb  + (size_t)GATES * HIDDEN;      // 16.78 MB in
    unsigned short* hb0   = wlinb + (size_t)VOCAB * HIDDEN;      // 119.7 MB in
    unsigned short* hb1   = hb0 + HIDDEN;
    float*          cst   = (float*)(hb1 + HIDDEN);
    unsigned long long* slots = (unsigned long long*)(cst + HIDDEN);

    // One-time bf16 weight conversion (keeps the whole model inside 192MB L2).
    cvt_bf16_kernel<<<1024, 256, 0, stream>>>(w_ih,  wihb,  (long long)(GATES * XDIM)  / 4);
    cvt_bf16_kernel<<<1024, 256, 0, stream>>>(w_hh,  whhb,  (long long)(GATES * HIDDEN) / 4);
    cvt_bf16_kernel<<<4096, 256, 0, stream>>>(w_lin, wlinb, ((long long)VOCAB * HIDDEN) / 4);
    init_state_kernel<<<4, 256, 0, stream>>>(h0, c0, hb0, cst, slots);

    for (int t = 0; t < SEQLEN; ++t) {
        unsigned short* hin  = (t & 1) ? hb1 : hb0;   // double-buffer h
        unsigned short* hout = (t & 1) ? hb0 : hb1;
        lstm_cell_kernel<<<64, 256, 0, stream>>>(wihb, whhb, seq, emb, hin, hout,
                                                 cst, b_ih, b_hh,
                                                 &slots[(t + 1) & 1], &slots[t & 1], t);
        logits_kernel<<<197, 256, 0, stream>>>(wlinb, hout, b_lin, &slots[t & 1],
                                               (t == SEQLEN - 1) ? (float*)d_out
                                                                 : (float*)nullptr);
    }
}
